// Model_1898375545583
// MI455X (gfx1250) — matrix operand order repaired and checked by reading, not compiled, not run on hardware
//
#include <hip/hip_runtime.h>
#include <hip/hip_bf16.h>
#include <math.h>

// ---------------------------------------------------------------------------
// Problem constants (from the reference)
//   D=256, LQ=16, NC=16, LC=12, NS=2048, LS=50 (padded to 64), TOPK=5
// ---------------------------------------------------------------------------
#define NEGV (-1e9f)

typedef __attribute__((ext_vector_type(16))) __bf16 v16bf;
typedef __attribute__((ext_vector_type(8)))  __bf16 v8bf;
typedef __attribute__((ext_vector_type(8)))  float  v8f;
typedef __attribute__((ext_vector_type(4)))  int    v4i32;

__device__ __forceinline__ __bf16 tobf(float x) { return (__bf16)x; }

// ---------------------------------------------------------------------------
// CDNA5 async global->LDS staging (ASYNCcnt path), with synchronous fallback
// Builtin prototype (per hipcc diagnostic): (int4 AS1*, int4 AS3*, imm, imm)
// ---------------------------------------------------------------------------
#if defined(__has_builtin)
#  if __has_builtin(__builtin_amdgcn_global_load_async_to_lds_b128) && \
      __has_builtin(__builtin_amdgcn_s_wait_asynccnt)
#    define HAVE_ASYNC_LDS 1
#  endif
#endif
#ifndef HAVE_ASYNC_LDS
#  define HAVE_ASYNC_LDS 0
#endif

__device__ __forceinline__ void lds_copy16B(__bf16* dst_lds, const __bf16* src_glob) {
#if HAVE_ASYNC_LDS
  __builtin_amdgcn_global_load_async_to_lds_b128(
      (__attribute__((address_space(1))) v4i32*)(void*)src_glob,
      (__attribute__((address_space(3))) v4i32*)(void*)dst_lds, 0, 0);
#else
  *(v8bf*)dst_lds = *(const v8bf*)src_glob;
#endif
}
__device__ __forceinline__ void lds_copy_wait() {
#if HAVE_ASYNC_LDS
  __builtin_amdgcn_s_wait_asynccnt(0);
#endif
}

// A-fragment (16x32 bf16) wave32, contiguous k order (same as frag_b):
//   lane = m(0..15) + 16*half ; elems -> K = kb + 16*half + e
__device__ __forceinline__ v16bf frag_a(const __bf16* A, int rs, int r0, int kb, int lane) {
  const int m = lane & 15, h = lane >> 4;
  return *(const v16bf*)(A + (size_t)(r0 + m) * rs + kb + 16 * h);
}

// B-fragment (32x16 bf16): lane = n(0..15) + 16*half ; elems -> K = kb + 16*half + e
// B is stored as the row-major "other matrix" S[n][k] (contraction dim contiguous).
__device__ __forceinline__ v16bf frag_b(const __bf16* B, int rs, int n0, int kb, int lane) {
  const int n = lane & 15, h = lane >> 4;
  return *(const v16bf*)(B + (size_t)(n0 + n) * rs + kb + 16 * h);
}

#define WMMA_BF16(a, b, c) \
  __builtin_amdgcn_wmma_f32_16x16x32_bf16(false, (a), false, (b), (short)0, (c), false, false)

// D/C layout: col = lane&15, row = r + 8*(lane>>4)

// ---------------------------------------------------------------------------
// block reductions (blockDim.x == 256)
// ---------------------------------------------------------------------------
__device__ __forceinline__ float redmax256(float v) {
  __shared__ float buf[256];
  int t = threadIdx.x;
  buf[t] = v; __syncthreads();
  for (int s = 128; s > 0; s >>= 1) { if (t < s) buf[t] = fmaxf(buf[t], buf[t + s]); __syncthreads(); }
  float r = buf[0]; __syncthreads(); return r;
}
__device__ __forceinline__ float redsum256(float v) {
  __shared__ float buf[256];
  int t = threadIdx.x;
  buf[t] = v; __syncthreads();
  for (int s = 128; s > 0; s >>= 1) { if (t < s) buf[t] += buf[t + s]; __syncthreads(); }
  float r = buf[0]; __syncthreads(); return r;
}

// ---------------------------------------------------------------------------
// k_tobf: one-time fp32 -> bf16 conversion of sents, zero-padded to 64 rows
// so per-sentence tiles can be async-copied blindly (no OOB, pad rows = 0).
// ---------------------------------------------------------------------------
__global__ void k_tobf(const float* sents, __bf16* sents_bf) {
  const int n = blockIdx.x, t = threadIdx.x;
  for (int idx = t; idx < 64 * 256; idx += 256) {
    int l = idx >> 8, d = idx & 255;
    sents_bf[(size_t)n * 64 * 256 + idx] =
        tobf((l < 50) ? sents[((size_t)n * 50 + l) * 256 + d] : 0.f);
  }
}

// ---------------------------------------------------------------------------
// k_prep: tiny softmaxes + bias dots + bf16 A-panels
// ---------------------------------------------------------------------------
__global__ void k_prep(
    const float* qvec, const float* cands,
    const float* Sqw, const float* Sqb, const float* Scw, const float* Scb,
    const float* wq1, const float* wq2, const float* wq3,
    const float* wc1, const float* wc2, const float* wc3,
    const float* fq1, const float* fq2, const float* fq3,
    const float* fc1, const float* fc2, const float* fc3,
    const int* cand_lens,
    __bf16* actx, __bf16* afq, __bf16* afc,
    float* self_q, float* self_c, float* a_q, float* a_c,
    float* mcfc1, float* qafq1, float* self_qvec, float* self_cvecs) {
  const int t = threadIdx.x;
  __shared__ float zq[16];
  __shared__ float zc[192];
  if (t < 16) {
    float z = Sqb[0], a = 0.f, f = 0.f;
    for (int d = 0; d < 256; ++d) {
      float v = qvec[t * 256 + d];
      z += v * Sqw[d]; a += v * wq1[d]; f += v * fq1[d];
    }
    zq[t] = z; a_q[t] = a; qafq1[t] = f;
  }
  if (t < 192) {
    int c = t / 12, i = t - 12 * c;
    float z = Scb[0], a = 0.f, f = 0.f;
    for (int d = 0; d < 256; ++d) {
      float v = cands[(size_t)t * 256 + d];
      z += v * Scw[d]; a += v * wc1[d]; f += v * fc1[d];
    }
    zc[t] = (i < cand_lens[c]) ? z : NEGV;
    a_c[t] = a; mcfc1[t] = f;
  }
  __syncthreads();
  if (t < 16) {
    float mx = -3.4e38f;
    for (int q = 0; q < 16; ++q) mx = fmaxf(mx, zq[q]);
    float sm = 0.f;
    for (int q = 0; q < 16; ++q) sm += expf(zq[q] - mx);
    self_q[t] = expf(zq[t] - mx) / sm;
  }
  if (t < 192) {
    int c = t / 12;
    float mx = -3.4e38f;
    for (int i = 0; i < 12; ++i) mx = fmaxf(mx, zc[c * 12 + i]);
    float sm = 0.f;
    for (int i = 0; i < 12; ++i) sm += expf(zc[c * 12 + i] - mx);
    self_c[t] = expf(zc[t] - mx) / sm;
  }
  __syncthreads();
  if (t < 256) {
    float s = 0.f;
    for (int q = 0; q < 16; ++q) s += self_q[q] * qvec[q * 256 + t];
    self_qvec[t] = s;
    for (int c = 0; c < 16; ++c) {
      float v = 0.f;
      for (int i = 0; i < 12; ++i) v += self_c[c * 12 + i] * cands[(size_t)(c * 12 + i) * 256 + t];
      self_cvecs[c * 256 + t] = v;
    }
  }
  // A panel for the context GEMM: rows 0..15 qvec*wq3, 16 wq2, 17 wc2, 18..31 zero,
  // rows 32..223 cand tokens * wc3
  for (int idx = t; idx < 224 * 256; idx += 256) {
    int r = idx >> 8, d = idx & 255;
    float v;
    if (r < 16)       v = qvec[r * 256 + d] * wq3[d];
    else if (r == 16) v = wq2[d];
    else if (r == 17) v = wc2[d];
    else if (r < 32)  v = 0.f;
    else              v = cands[(size_t)(r - 32) * 256 + d] * wc3[d];
    actx[idx] = tobf(v);
  }
  // find-q panel: rows 0..15 qvec*fq3, row 16 = fq2
  for (int idx = t; idx < 32 * 256; idx += 256) {
    int r = idx >> 8, d = idx & 255;
    float v = (r < 16) ? qvec[r * 256 + d] * fq3[d] : ((r == 16) ? fq2[d] : 0.f);
    afq[idx] = tobf(v);
  }
  // find-c panels (per c): rows 0..11 cands[c]*fc3, row 12 = fc2
  for (int idx = t; idx < 16 * 16 * 256; idx += 256) {
    int d = idx & 255, rc = idx >> 8;
    int c = rc >> 4, r = rc & 15;
    float v = (r < 12) ? cands[(size_t)(c * 12 + r) * 256 + d] * fc3[d]
                       : ((r == 12) ? fc2[d] : 0.f);
    afc[idx] = tobf(v);
  }
}

// ---------------------------------------------------------------------------
// k_ctx: one block per sentence n. Async-stage bf16 tile, WMMA logits
// (224 x 64) fused with masked softmax over l; emits Hscore_q, Hq, Hc.
// ---------------------------------------------------------------------------
__global__ __launch_bounds__(128) void k_ctx(
    const __bf16* sents_bf, const int* sent_lens, const __bf16* actx,
    const float* self_q, const float* self_c, const float* a_q, const float* a_c,
    float* hsq, float* hq, __bf16* hq_bf, __bf16* hc_bf) {
  const int n = blockIdx.x, t = threadIdx.x;
  const int lane = t & 31, wv = t >> 5;
  const int len = sent_lens[n];
  __shared__ alignas(32) __bf16 S[64 * 256];
  __shared__ float LG[16 * 64];
  __shared__ float TQ[64], TC[64], HS[64];
  __shared__ float WC[16 * 64];

  // async stage the 32 KB bf16 sentence tile (2048 x 16B chunks)
  {
    const __bf16* gsrc = sents_bf + (size_t)n * 64 * 256;
    for (int cid = t; cid < 2048; cid += 128) lds_copy16B(S + cid * 8, gsrc + cid * 8);
    lds_copy_wait();
  }
  for (int idx = t; idx < 16 * 64; idx += 128) WC[idx] = 0.f;
  __syncthreads();

  for (int ii = 0; ii < 14; ++ii) {
    const int mt = (ii == 0) ? 1 : ((ii == 1) ? 0 : ii);  // do the bias-row tile first
    v8f acc = {};
#pragma unroll
    for (int kb = 0; kb < 8; ++kb) {
      v16bf a = frag_a(actx, 256, mt * 16, kb * 32, lane);
      v16bf b = frag_b(S, 256, wv * 16, kb * 32, lane);
      acc = WMMA_BF16(a, b, acc);
    }
    {
      int h = lane >> 4, col = (lane & 15) + wv * 16;
#pragma unroll
      for (int r = 0; r < 8; ++r) LG[(r + 8 * h) * 64 + col] = acc[r];
    }
    __syncthreads();
    if (mt == 1) {
      if (t < 64) { TQ[t] = LG[0 * 64 + t]; TC[t] = LG[1 * 64 + t]; }
    } else {
      if (t < 16) {
        float base = (mt == 0) ? a_q[t] : a_c[(mt - 2) * 16 + t];
        const float* TT = (mt == 0) ? TQ : TC;
        float mx = -3.4e38f;
        for (int l = 0; l < 50; ++l) {
          float v = (l < len) ? base + TT[l] + LG[t * 64 + l] : NEGV;
          mx = fmaxf(mx, v);
        }
        float sm = 0.f;
        for (int l = 0; l < 50; ++l) {
          float v = (l < len) ? base + TT[l] + LG[t * 64 + l] : NEGV;
          sm += expf(v - mx);
        }
        float inv = 1.f / sm;
        for (int l = 0; l < 50; ++l) {
          float v = (l < len) ? base + TT[l] + LG[t * 64 + l] : NEGV;
          LG[t * 64 + l] = expf(v - mx) * inv;
        }
      }
      __syncthreads();
      if (t < 50) {
        int l = t;
        if (mt == 0) {
          float s = 0.f;
          for (int q = 0; q < 16; ++q) s += self_q[q] * LG[q * 64 + l];
          HS[l] = s;
        } else {
          for (int r = 0; r < 16; ++r) {
            int m = (mt - 2) * 16 + r;
            int c = m / 12, i = m - 12 * c;
            WC[c * 64 + l] += self_c[c * 12 + i] * LG[r * 64 + l];
          }
        }
      }
    }
    __syncthreads();
  }
  if (t < 64) hsq[(size_t)n * 64 + t] = (t < 50) ? HS[t] : 0.f;
  for (int d = t; d < 256; d += 128) {
    float h = 0.f;
    for (int l = 0; l < 50; ++l) h += HS[l] * (float)S[l * 256 + d];
    hq[(size_t)n * 256 + d] = h;
    hq_bf[(size_t)n * 256 + d] = tobf(h);
  }
  for (int c = 0; c < 16; ++c) {
    for (int d = t; d < 256; d += 128) {
      float v = 0.f;
      for (int l = 0; l < 50; ++l) v += WC[c * 64 + l] * (float)S[l * 256 + d];
      hc_bf[((size_t)c * 2048 + n) * 256 + d] = tobf(v);
    }
  }
}

// ---------------------------------------------------------------------------
// k_selfatt: per-sentence 64x64 WMMA attention, fused kk-masked softmax.
// Bilinear term re-associated: qpk_i . (s_j*w3) = ((q+s_i)*w3) . s_j, so the
// B operand is the raw async-staged tile. Used twice (ws* then wm*).
// ---------------------------------------------------------------------------
__global__ __launch_bounds__(128) void k_selfatt(
    const __bf16* sents_bf, const int* sent_lens,
    const float* w1, const float* w2, const float* w3,
    const float* query, const float* att,
    float* outSC, float* outH) {
  const int n = blockIdx.x, t = threadIdx.x;
  const int lane = t & 31, wv = t >> 5;
  const int len = sent_lens[n];
  __shared__ alignas(32) __bf16 S[64 * 256];
  __shared__ alignas(32) __bf16 QP3[64 * 256];
  __shared__ float LG[64 * 64];
  __shared__ float Qv[256];
  __shared__ float A1[64], B2[64], CS[64];

  __builtin_prefetch(sents_bf + (size_t)n * 64 * 256, 0, 1);

  {
    const __bf16* gsrc = sents_bf + (size_t)n * 64 * 256;
    for (int cid = t; cid < 2048; cid += 128) lds_copy16B(S + cid * 8, gsrc + cid * 8);
    lds_copy_wait();
  }
  for (int d = t; d < 256; d += 128) Qv[d] = query[(size_t)n * 256 + d];
  __syncthreads();

  // A panel: (q + s_i) * w3 (pad rows produce unused logit rows)
  for (int idx = t; idx < 64 * 256; idx += 128) {
    int d = idx & 255;
    QP3[idx] = tobf((Qv[d] + (float)S[idx]) * w3[d]);
  }
  if (t < 64) {
    float a = 0.f, b = 0.f;
    for (int d = 0; d < 256; ++d) {
      float s = (float)S[t * 256 + d];
      a += (Qv[d] + s) * w1[d];
      b += s * w2[d];
    }
    A1[t] = a; B2[t] = b;
  }
  __syncthreads();
  for (int mt = 0; mt < 4; ++mt) {
    v8f acc = {};
#pragma unroll
    for (int kb = 0; kb < 8; ++kb) {
      v16bf a = frag_a(QP3, 256, mt * 16, kb * 32, lane);
      v16bf b = frag_b(S, 256, wv * 16, kb * 32, lane);
      acc = WMMA_BF16(a, b, acc);
    }
    int h = lane >> 4, col = (lane & 15) + wv * 16;
#pragma unroll
    for (int r = 0; r < 8; ++r) LG[(mt * 16 + r + 8 * h) * 64 + col] = acc[r];
  }
  __syncthreads();
  if (t < 50) {
    const int i = t;
    float mx = -3.4e38f;
    for (int j = 0; j < 50; ++j) {
      float v = (j < len && i < len) ? A1[i] + B2[j] + LG[i * 64 + j] : NEGV;
      mx = fmaxf(mx, v);
    }
    float sm = 0.f;
    for (int j = 0; j < 50; ++j) {
      float v = (j < len && i < len) ? A1[i] + B2[j] + LG[i * 64 + j] : NEGV;
      sm += expf(v - mx);
    }
    float sc = att[(size_t)n * 64 + i] / sm;
    for (int j = 0; j < 50; ++j) {
      float v = (j < len && i < len) ? A1[i] + B2[j] + LG[i * 64 + j] : NEGV;
      LG[i * 64 + j] = expf(v - mx) * sc;
    }
  }
  __syncthreads();
  if (t < 64) {
    float s = 0.f;
    if (t < 50) for (int i = 0; i < 50; ++i) s += LG[i * 64 + t];
    CS[t] = s;
    outSC[(size_t)n * 64 + t] = s;
  }
  __syncthreads();
  for (int d = t; d < 256; d += 128) {
    float h = 0.f;
    for (int j = 0; j < 50; ++j) h += CS[j] * (float)S[j * 256 + d];
    outH[(size_t)n * 256 + d] = h;
  }
}

// ---------------------------------------------------------------------------
// find-q GEMM: [32 x 2048] over K=256 (row 16 is fq2 -> hb[n])
// ---------------------------------------------------------------------------
__global__ __launch_bounds__(32) void k_findq_gemm(
    const __bf16* afq, const __bf16* hq_bf, float* sqraw, float* hb) {
  const int lane = threadIdx.x;
  const int n0 = blockIdx.x * 16;
  v8f acc0 = {}; v8f acc1 = {};
#pragma unroll
  for (int kb = 0; kb < 8; ++kb) {
    v16bf b  = frag_b(hq_bf, 256, n0, kb * 32, lane);
    v16bf a0 = frag_a(afq, 256, 0,  kb * 32, lane);
    v16bf a1 = frag_a(afq, 256, 16, kb * 32, lane);
    acc0 = WMMA_BF16(a0, b, acc0);
    acc1 = WMMA_BF16(a1, b, acc1);
  }
  const int h = lane >> 4, col = n0 + (lane & 15);
#pragma unroll
  for (int r = 0; r < 8; ++r) sqraw[(size_t)(r + 8 * h) * 2048 + col] = acc0[r];
  if (lane < 16) hb[col] = acc1[0];  // tile-1 row 0 == global row 16 (fq2)
}

__global__ void k_rowsoft_q(const float* sqraw, const float* hb, const float* qafq1, float* smq) {
  const int q = blockIdx.x, t = threadIdx.x;
  const float c = qafq1[q];
  float mx = -3.4e38f;
  for (int n = t; n < 2048; n += 256) mx = fmaxf(mx, sqraw[q * 2048 + n] + hb[n] + c);
  mx = redmax256(mx);
  float sm = 0.f;
  for (int n = t; n < 2048; n += 256) sm += expf(sqraw[q * 2048 + n] + hb[n] + c - mx);
  sm = redsum256(sm);
  const float inv = 1.f / sm;
  for (int n = t; n < 2048; n += 256) smq[q * 2048 + n] = expf(sqraw[q * 2048 + n] + hb[n] + c - mx) * inv;
}

__global__ void k_q2patt(const float* self_q, const float* smq, const float* hq,
                         float* sqv, float* q2patt) {
  const int t = threadIdx.x;
  for (int n = t; n < 2048; n += 256) {
    float s = 0.f;
    for (int q = 0; q < 16; ++q) s += self_q[q] * smq[q * 2048 + n];
    sqv[n] = s;
  }
  __syncthreads();
  float a = 0.f;
  for (int n = 0; n < 2048; ++n) a += sqv[n] * hq[(size_t)n * 256 + t];
  q2patt[t] = a;
}

// ---------------------------------------------------------------------------
// find-c GEMM: per c, [16 x 2048] over K=256 (row 12 is fc2)
// ---------------------------------------------------------------------------
__global__ __launch_bounds__(32) void k_findc_gemm(
    const __bf16* afc, const __bf16* hc_bf, float* scr) {
  const int lane = threadIdx.x;
  const int n0 = blockIdx.x * 16, c = blockIdx.y;
  const __bf16* A = afc + (size_t)c * 16 * 256;
  const __bf16* B = hc_bf + (size_t)c * 2048 * 256;
  v8f acc = {};
#pragma unroll
  for (int kb = 0; kb < 8; ++kb) {
    v16bf a = frag_a(A, 256, 0, kb * 32, lane);
    v16bf b = frag_b(B, 256, n0, kb * 32, lane);
    acc = WMMA_BF16(a, b, acc);
  }
  const int h = lane >> 4, col = n0 + (lane & 15);
#pragma unroll
  for (int r = 0; r < 8; ++r) scr[((size_t)c * 16 + r + 8 * h) * 2048 + col] = acc[r];
}

__global__ void k_findc_soft(const float* scr, const float* mcfc1, const int* cand_lens, float* smc) {
  const int b = blockIdx.x, t = threadIdx.x;
  const int c = b / 12, l = b - 12 * c;
  const bool valid = l < cand_lens[c];
  const float* rowl  = scr + ((size_t)c * 16 + l) * 2048;
  const float* row12 = scr + ((size_t)c * 16 + 12) * 2048;
  const float base = mcfc1[c * 12 + l];
  float mx = -3.4e38f;
  for (int n = t; n < 2048; n += 256) { float v = valid ? base + row12[n] + rowl[n] : NEGV; mx = fmaxf(mx, v); }
  mx = redmax256(mx);
  float sm = 0.f;
  for (int n = t; n < 2048; n += 256) { float v = valid ? base + row12[n] + rowl[n] : NEGV; sm += expf(v - mx); }
  sm = redsum256(sm);
  const float inv = 1.f / sm;
  for (int n = t; n < 2048; n += 256) {
    float v = valid ? base + row12[n] + rowl[n] : NEGV;
    smc[((size_t)c * 12 + l) * 2048 + n] = expf(v - mx) * inv;
  }
}

__global__ void k_c2p(const float* smc, const float* self_c, const __bf16* hc_bf, float* c2patt) {
  const int c = blockIdx.x, t = threadIdx.x;
  __shared__ float comb[2048];
  for (int n = t; n < 2048; n += 256) {
    float s = 0.f;
    for (int l = 0; l < 12; ++l) s += self_c[c * 12 + l] * smc[((size_t)c * 12 + l) * 2048 + n];
    comb[n] = s;
  }
  __syncthreads();
  float a = 0.f;
  const __bf16* H = hc_bf + (size_t)c * 2048 * 256;
  for (int n = 0; n < 2048; ++n) a += comb[n] * (float)H[(size_t)n * 256 + t];
  c2patt[c * 256 + t] = a;
}

// ---------------------------------------------------------------------------
// q2p / c2p composition: concat(sv, at, sv*at) @ Wcom_w + b
// ---------------------------------------------------------------------------
__global__ void k_compose(const float* self_qvec, const float* q2patt,
                          const float* self_cvecs, const float* c2patt,
                          const float* W, const float* bias,
                          float* out, float* q2pm) {
  const int b = blockIdx.x, d = threadIdx.x;
  const float *sv, *at; float* dst;
  if (b == 0) { sv = self_qvec; at = q2patt; dst = out; }
  else { sv = self_cvecs + (b - 1) * 256; at = c2patt + (b - 1) * 256; dst = out + 256 + (size_t)(b - 1) * 256; }
  float acc = bias[d];
  for (int j = 0; j < 256; ++j) acc += sv[j] * W[(size_t)j * 256 + d];
  for (int j = 0; j < 256; ++j) acc += at[j] * W[(size_t)(256 + j) * 256 + d];
  for (int j = 0; j < 256; ++j) acc += sv[j] * at[j] * W[(size_t)(512 + j) * 256 + d];
  dst[d] = acc;
  if (b == 0) q2pm[d] = acc;
}

// ---------------------------------------------------------------------------
// top-k chain
// ---------------------------------------------------------------------------
__global__ void k_s1(const float* hm, const float* q2pm, const float* wfindm, float* s1raw) {
  const int n = blockIdx.x * 256 + threadIdx.x;
  const float* wa = wfindm; const float* wb = wfindm + 256; const float* wc = wfindm + 512;
  float acc = 0.f;
  for (int d = 0; d < 256; ++d)
    acc += q2pm[d] * wa[d] + hm[(size_t)n * 256 + d] * (wb[d] + q2pm[d] * wc[d]);
  s1raw[n] = acc;
}

__global__ void k_s1top(const float* s1raw, const float* hm, float* s1,
                        float* score1w, float* v1w, float* out) {
  const int t = threadIdx.x;
  float mx = -3.4e38f;
  for (int n = t; n < 2048; n += 256) mx = fmaxf(mx, s1raw[n]);
  mx = redmax256(mx);
  float sm = 0.f;
  for (int n = t; n < 2048; n += 256) sm += expf(s1raw[n] - mx);
  sm = redsum256(sm);
  const float inv = 1.f / sm;
  for (int n = t; n < 2048; n += 256) s1[n] = expf(s1raw[n] - mx) * inv;
  __syncthreads();
  __shared__ int sind[5];
  if (t == 0) {
    for (int kk = 0; kk < 5; ++kk) {
      float bv = -3.4e38f; int bi = 0;
      for (int n = 0; n < 2048; ++n) if (s1[n] > bv) { bv = s1[n]; bi = n; }
      sind[kk] = bi; s1[bi] = -1.f;
      out[5632 + kk] = bv; score1w[kk] = bv;
    }
  }
  __syncthreads();
  for (int kk = 0; kk < 5; ++kk) {
    const float v = hm[(size_t)sind[kk] * 256 + t];
    out[4352 + kk * 256 + t] = v;   // v1 (3rd output)
    out[5637 + kk * 256 + t] = v;   // v1 again (5th output)
    v1w[kk * 256 + t] = v;
  }
}

__global__ void k_s2raw(const float* hm2, const float* v1w, const float* wfindm2, float* s2raw) {
  const int idx = blockIdx.x * 256 + threadIdx.x;
  const int k = idx >> 11, n = idx & 2047;
  const float* wa = wfindm2; const float* wb = wfindm2 + 256; const float* wc = wfindm2 + 512;
  const float* v = v1w + k * 256;
  float acc = 0.f;
  for (int d = 0; d < 256; ++d)
    acc += v[d] * wa[d] + hm2[(size_t)n * 256 + d] * (wb[d] + v[d] * wc[d]);
  s2raw[idx] = acc;
}

__global__ void k_s2top(float* s2raw, const float* score1w, const float* hm2, float* out) {
  const int t = threadIdx.x;
  __shared__ int sind[5];
  for (int k = 0; k < 5; ++k) {
    float* row = s2raw + (size_t)k * 2048;
    float mx = -3.4e38f;
    for (int n = t; n < 2048; n += 256) mx = fmaxf(mx, row[n]);
    mx = redmax256(mx);
    float sm = 0.f;
    for (int n = t; n < 2048; n += 256) sm += expf(row[n] - mx);
    sm = redsum256(sm);
    const float sc = score1w[k] / sm;
    for (int n = t; n < 2048; n += 256) row[n] = expf(row[n] - mx) * sc;
    __syncthreads();
    if (t == 0) {
      for (int j = 0; j < 5; ++j) {
        float bv = -3.4e38f; int bi = 0;
        for (int n = 0; n < 2048; ++n) if (row[n] > bv) { bv = row[n]; bi = n; }
        sind[j] = bi; row[bi] = -1e30f;
        out[13317 + k * 5 + j] = bv;
      }
    }
    __syncthreads();
    for (int j = 0; j < 5; ++j)
      out[6917 + (k * 5 + j) * 256 + t] = hm2[(size_t)sind[j] * 256 + t];
    __syncthreads();
  }
}

// ---------------------------------------------------------------------------
// launch
// ---------------------------------------------------------------------------
extern "C" void kernel_launch(void* const* d_in, const int* in_sizes, int n_in,
                              void* d_out, int out_size, void* d_ws, size_t ws_size,
                              hipStream_t stream) {
  (void)in_sizes; (void)n_in; (void)out_size; (void)ws_size;
  const float* qvec  = (const float*)d_in[0];
  const float* cands = (const float*)d_in[1];
  const float* sents = (const float*)d_in[2];
  const float* Sqw = (const float*)d_in[3];
  const float* Sqb = (const float*)d_in[4];
  const float* Scw = (const float*)d_in[5];
  const float* Scb = (const float*)d_in[6];
  const float *wq1 = (const float*)d_in[7],  *wq2 = (const float*)d_in[8],  *wq3 = (const float*)d_in[9];
  const float *wc1 = (const float*)d_in[10], *wc2 = (const float*)d_in[11], *wc3 = (const float*)d_in[12];
  const float *ws1 = (const float*)d_in[13], *ws2 = (const float*)d_in[14], *ws3 = (const float*)d_in[15];
  const float *wm1 = (const float*)d_in[16], *wm2 = (const float*)d_in[17], *wm3 = (const float*)d_in[18];
  const float *fq1 = (const float*)d_in[19], *fq2 = (const float*)d_in[20], *fq3 = (const float*)d_in[21];
  const float *fc1 = (const float*)d_in[22], *fc2 = (const float*)d_in[23], *fc3 = (const float*)d_in[24];
  const float* wfindm  = (const float*)d_in[25];
  const float* wfindm2 = (const float*)d_in[26];
  const float* Wcom_w  = (const float*)d_in[27];
  const float* Wcom_b  = (const float*)d_in[28];
  const int* cand_lens = (const int*)d_in[29];
  const int* sent_lens = (const int*)d_in[30];
  float* out = (float*)d_out;

  char* base = (char*)d_ws; size_t off = 0;
  auto wsalloc = [&](size_t bytes) -> void* {
    void* p = base + off; off = (off + bytes + 255) & ~(size_t)255; return p;
  };
  __bf16* sbf  = (__bf16*)wsalloc((size_t)2048 * 64 * 256 * 2);  // padded bf16 sents
  __bf16* actx = (__bf16*)wsalloc((size_t)224 * 256 * 2);
  __bf16* afq  = (__bf16*)wsalloc((size_t)32 * 256 * 2);
  __bf16* afc  = (__bf16*)wsalloc((size_t)16 * 16 * 256 * 2);
  __bf16* hqbf = (__bf16*)wsalloc((size_t)2048 * 256 * 2);
  __bf16* hcbf = (__bf16*)wsalloc((size_t)16 * 2048 * 256 * 2);
  float* self_q    = (float*)wsalloc(16 * 4);
  float* self_c    = (float*)wsalloc(192 * 4);
  float* a_q       = (float*)wsalloc(16 * 4);
  float* a_c       = (float*)wsalloc(192 * 4);
  float* mcfc1     = (float*)wsalloc(192 * 4);
  float* qafq1     = (float*)wsalloc(16 * 4);
  float* self_qvec = (float*)wsalloc(256 * 4);
  float* self_cvecs= (float*)wsalloc((size_t)16 * 256 * 4);
  float* hsq  = (float*)wsalloc((size_t)2048 * 64 * 4);
  float* hq   = (float*)wsalloc((size_t)2048 * 256 * 4);
  float* sc1  = (float*)wsalloc((size_t)2048 * 64 * 4);
  float* hm   = (float*)wsalloc((size_t)2048 * 256 * 4);
  float* sc2  = (float*)wsalloc((size_t)2048 * 64 * 4);
  float* hm2  = (float*)wsalloc((size_t)2048 * 256 * 4);
  float* sqraw= (float*)wsalloc((size_t)16 * 2048 * 4);
  float* hb   = (float*)wsalloc(2048 * 4);
  float* smq  = (float*)wsalloc((size_t)16 * 2048 * 4);
  float* sqv  = (float*)wsalloc(2048 * 4);
  float* q2patt = (float*)wsalloc(256 * 4);
  float* scr  = (float*)wsalloc((size_t)16 * 16 * 2048 * 4);
  float* smc  = (float*)wsalloc((size_t)16 * 12 * 2048 * 4);
  float* c2patt = (float*)wsalloc((size_t)16 * 256 * 4);
  float* q2pm = (float*)wsalloc(256 * 4);
  float* s1raw= (float*)wsalloc(2048 * 4);
  float* s1   = (float*)wsalloc(2048 * 4);
  float* s2raw= (float*)wsalloc((size_t)5 * 2048 * 4);
  float* score1w = (float*)wsalloc(5 * 4);
  float* v1w  = (float*)wsalloc((size_t)5 * 256 * 4);

  k_tobf<<<2048, 256, 0, stream>>>(sents, sbf);
  k_prep<<<1, 256, 0, stream>>>(qvec, cands, Sqw, Sqb, Scw, Scb,
                                wq1, wq2, wq3, wc1, wc2, wc3,
                                fq1, fq2, fq3, fc1, fc2, fc3,
                                cand_lens, actx, afq, afc,
                                self_q, self_c, a_q, a_c, mcfc1, qafq1,
                                self_qvec, self_cvecs);
  k_ctx<<<2048, 128, 0, stream>>>(sbf, sent_lens, actx, self_q, self_c,
                                  a_q, a_c, hsq, hq, hqbf, hcbf);
  k_selfatt<<<2048, 128, 0, stream>>>(sbf, sent_lens, ws1, ws2, ws3, hq, hsq, sc1, hm);
  k_selfatt<<<2048, 128, 0, stream>>>(sbf, sent_lens, wm1, wm2, wm3, hm, sc1, sc2, hm2);
  k_findq_gemm<<<128, 32, 0, stream>>>(afq, hqbf, sqraw, hb);
  k_rowsoft_q<<<16, 256, 0, stream>>>(sqraw, hb, qafq1, smq);
  k_q2patt<<<1, 256, 0, stream>>>(self_q, smq, hq, sqv, q2patt);
  k_findc_gemm<<<dim3(128, 16), 32, 0, stream>>>(afc, hcbf, scr);
  k_findc_soft<<<192, 256, 0, stream>>>(scr, mcfc1, cand_lens, smc);
  k_c2p<<<16, 256, 0, stream>>>(smc, self_c, hcbf, c2patt);
  k_compose<<<17, 256, 0, stream>>>(self_qvec, q2patt, self_cvecs, c2patt,
                                    Wcom_w, Wcom_b, out, q2pm);
  k_s1<<<8, 256, 0, stream>>>(hm, q2pm, wfindm, s1raw);
  k_s1top<<<1, 256, 0, stream>>>(s1raw, hm, s1, score1w, v1w, out);
  k_s2raw<<<40, 256, 0, stream>>>(hm2, v1w, wfindm2, s2raw);
  k_s2top<<<1, 256, 0, stream>>>(s2raw, score1w, hm2, out);
}
